// WMIL_15685220565499
// MI455X (gfx1250) — matrix operand order repaired and checked by reading, not compiled, not run on hardware
//
#include <hip/hip_runtime.h>
#include <hip/hip_bf16.h>

typedef __attribute__((ext_vector_type(16))) __bf16 v16bf;
typedef __attribute__((ext_vector_type(8)))  float  v8f;
typedef __attribute__((ext_vector_type(4)))  int    v4i;

#define B_BAGS   8
#define N_INST   50000
#define D_IN     1024
#define H_DIM    128
#define N_TILES  3125                    // N_INST / 16 (exact)
#define TPW      2                       // row tiles per wave
#define WAVES_PER_BAG ((N_TILES + TPW - 1) / TPW)   // 1563
#define K_STEPS  (D_IN / 32)             // 32
#define GEMM_BLOCKS ((WAVES_PER_BAG + 7) / 8)       // 8 waves / block
#define KSTEP_BYTES 16384                // 16 fragments x 1KB per k-step

#if defined(__has_builtin)
#  if __has_builtin(__builtin_amdgcn_global_load_async_to_lds_b128)
#    define HAVE_ASYNC_LDS 1
#  else
#    define HAVE_ASYNC_LDS 0
#  endif
#  if __has_builtin(__builtin_amdgcn_s_wait_asynccnt)
#    define HAVE_WAIT_ASYNC 1
#  else
#    define HAVE_WAIT_ASYNC 0
#  endif
#else
#  define HAVE_ASYNC_LDS 0
#  define HAVE_WAIT_ASYNC 0
#endif

// ---------------------------------------------------------------------------
// Kernel 0: prepack Wv/Wu (fp32 [1024,128] row-major) into bf16 fragments in
// the 16x16x32 B-matrix layout used here (K contiguous per lane half, same as A):
//   lanes 0-15 : K = kbase+0..15 (dword j holds K=2j,2j+1), col N = lane%16
//   lanes 16-31: K = kbase+16..31
// Fragment id f = kstep*16 + mat*8 + htile  (=> one contiguous 16KB block per
// k-step, so a workgroup can stage it into LDS with one coalesced sweep).
// ---------------------------------------------------------------------------
__global__ __launch_bounds__(256) void wmil_prepack(
    const float* __restrict__ Wv, const float* __restrict__ Wu,
    unsigned short* __restrict__ wpack)
{
    int t = blockIdx.x * 256 + threadIdx.x;          // 0 .. 16383
    int lane  = t & 31;
    int kstep = (t >> 5) & 31;
    int h     = (t >> 10) & 7;
    int m     = (t >> 13) & 1;
    const float* W = m ? Wu : Wv;
    int n  = h * 16 + (lane & 15);
    int kb = kstep * 32 + ((lane < 16) ? 0 : 16);
    unsigned int dw[8];
#pragma unroll
    for (int j = 0; j < 8; ++j) {
        float lo = W[(size_t)(kb + 2 * j)     * H_DIM + n];
        float hi = W[(size_t)(kb + 2 * j + 1) * H_DIM + n];
        __bf16 bl = (__bf16)lo;
        __bf16 bh = (__bf16)hi;
        unsigned short ls = __builtin_bit_cast(unsigned short, bl);
        unsigned short hs = __builtin_bit_cast(unsigned short, bh);
        dw[j] = (unsigned int)ls | ((unsigned int)hs << 16);
    }
    int f = kstep * 16 + m * 8 + h;                  // 0..511
    unsigned int* dst = (unsigned int*)(wpack + (size_t)f * 512 + lane * 16);
    *(uint4*)(dst)     = make_uint4(dw[0], dw[1], dw[2], dw[3]);
    *(uint4*)(dst + 4) = make_uint4(dw[4], dw[5], dw[6], dw[7]);
}

// ---------------------------------------------------------------------------
// Kernel 1: fused dual GEMM (x@Wv, x@Wu) + tanh*sigmoid gate + score = A@Ww+bw
// One wave handles TPW 16-row tiles; 8 H-tiles of f32 accumulators each.
// Per k-step, the workgroup stages the 16KB of B fragments into a
// double-buffered LDS tile via GLOBAL_LOAD_ASYNC_TO_LDS_B128 (ASYNCcnt).
// ---------------------------------------------------------------------------
__global__ __launch_bounds__(256) void wmil_gemm_scores(
    const float* __restrict__ x,
    const unsigned short* __restrict__ wpack,
    const float* __restrict__ bv, const float* __restrict__ bu,
    const float* __restrict__ Ww, const float* __restrict__ bw,
    float* __restrict__ scores)
{
    __shared__ __align__(32) unsigned char smem[2][KSTEP_BYTES];

    const int tid  = threadIdx.x;
    const int lane = tid & 31;
    const int wid  = (blockIdx.x * 256 + tid) >> 5;  // wave id within bag
    const int bag  = blockIdx.y;
    const int tile0 = wid * TPW;                     // may exceed N_TILES; clamp

    const int nlo  = lane & 15;
    const int koff = (lane < 16) ? 0 : 16;           // A: K = 16*half + s (matches B)

    float bvv[8], buu[8], www[8];
#pragma unroll
    for (int h = 0; h < 8; ++h) {
        int n = h * 16 + nlo;
        bvv[h] = bv[n]; buu[h] = bu[n]; www[h] = Ww[n];
    }
    const float bias_w = bw[0];

    const float* xrow[TPW];
    int tidx[TPW];
#pragma unroll
    for (int t = 0; t < TPW; ++t) {
        int tt = tile0 + t; if (tt > N_TILES - 1) tt = N_TILES - 1;
        tidx[t] = tt;
        xrow[t] = x + ((size_t)bag * N_INST + (size_t)tt * 16 + nlo) * D_IN;
    }

    const unsigned char* wsrc = (const unsigned char*)wpack;

    // --- cooperative stage of one k-step (16KB) into smem[buf]; 64B/thread ---
    auto stage = [&](int ks, int buf) {
        const unsigned char* g = wsrc + (size_t)ks * KSTEP_BYTES + (size_t)tid * 64;
        unsigned char* l = &smem[buf][tid * 64];
#if HAVE_ASYNC_LDS
        typedef __attribute__((address_space(1))) v4i g4;   // global int4
        typedef __attribute__((address_space(3))) v4i l4;   // LDS int4
#pragma unroll
        for (int q = 0; q < 4; ++q)
            __builtin_amdgcn_global_load_async_to_lds_b128(
                (g4*)(g + 16 * q), (l4*)(l + 16 * q), 0, 0);
#else
        const uint4* gs = (const uint4*)g;
        uint4* ld = (uint4*)l;
#pragma unroll
        for (int q = 0; q < 4; ++q) ld[q] = gs[q];
#endif
    };
    auto stage_wait = [&]() {
#if HAVE_ASYNC_LDS
#  if HAVE_WAIT_ASYNC
        __builtin_amdgcn_s_wait_asynccnt(0);
#  else
        asm volatile("s_wait_asynccnt 0x0" ::: "memory");
#  endif
#endif
    };

    const v8f vzero = {0.f, 0.f, 0.f, 0.f, 0.f, 0.f, 0.f, 0.f};
    v8f accV[TPW][8];
    v8f accU[TPW][8];
#pragma unroll
    for (int t = 0; t < TPW; ++t)
#pragma unroll
        for (int h = 0; h < 8; ++h) { accV[t][h] = vzero; accU[t][h] = vzero; }

    stage(0, 0);
    stage_wait();

    for (int ks = 0; ks < K_STEPS; ++ks) {
        const int cur = ks & 1;
        __syncthreads();                 // buf[cur] staged by all; buf[1-cur] free
        if (ks + 1 < K_STEPS) stage(ks + 1, 1 - cur);

        v16bf afr[TPW];
#pragma unroll
        for (int t = 0; t < TPW; ++t) {
            const float* p = xrow[t] + ks * 32 + koff;
            float4 f0 = *(const float4*)(p);
            float4 f1 = *(const float4*)(p + 4);
            float4 f2 = *(const float4*)(p + 8);
            float4 f3 = *(const float4*)(p + 12);
            v16bf a;
            a[0]  = (__bf16)f0.x; a[1]  = (__bf16)f0.y; a[2]  = (__bf16)f0.z; a[3]  = (__bf16)f0.w;
            a[4]  = (__bf16)f1.x; a[5]  = (__bf16)f1.y; a[6]  = (__bf16)f1.z; a[7]  = (__bf16)f1.w;
            a[8]  = (__bf16)f2.x; a[9]  = (__bf16)f2.y; a[10] = (__bf16)f2.z; a[11] = (__bf16)f2.w;
            a[12] = (__bf16)f3.x; a[13] = (__bf16)f3.y; a[14] = (__bf16)f3.z; a[15] = (__bf16)f3.w;
            afr[t] = a;
        }

        const v16bf* lf = (const v16bf*)smem[cur];   // 16 frags x 32 lanes
#pragma unroll
        for (int h = 0; h < 8; ++h) {
            v16bf bV = lf[(0 * 8 + h) * 32 + lane];
            v16bf bU = lf[(1 * 8 + h) * 32 + lane];
#pragma unroll
            for (int t = 0; t < TPW; ++t) {
                accV[t][h] = __builtin_amdgcn_wmma_f32_16x16x32_bf16(
                    false, afr[t], false, bV, (short)0, accV[t][h], false, false);
                accU[t][h] = __builtin_amdgcn_wmma_f32_16x16x32_bf16(
                    false, afr[t], false, bU, (short)0, accU[t][h], false, false);
            }
        }
        if (ks + 1 < K_STEPS) stage_wait();          // own async writes landed
    }

    // Epilogue: gate, dot with Ww, reduce 16 lanes per half, store score.
#pragma unroll
    for (int t = 0; t < TPW; ++t) {
#pragma unroll
        for (int r = 0; r < 8; ++r) {
            float p = 0.f;
#pragma unroll
            for (int h = 0; h < 8; ++h) {
                float av = accV[t][h][r] + bvv[h];
                float au = accU[t][h][r] + buu[h];
                float tv = tanhf(av);
                float sg = 1.0f / (1.0f + __expf(-au));
                p += tv * sg * www[h];
            }
            p += __shfl_xor(p, 1, 32);
            p += __shfl_xor(p, 2, 32);
            p += __shfl_xor(p, 4, 32);
            p += __shfl_xor(p, 8, 32);
            float sc = p + bias_w;                   // TEMPERATURE == 1
            int row = tidx[t] * 16 + ((lane < 16) ? r : (8 + r));
            if (nlo == 0 && (tile0 + t) < N_TILES)
                scores[(size_t)bag * N_INST + row] = sc;
        }
    }
}

// ---------------------------------------------------------------------------
// Kernel 2: per-bag top-10 + bottom-10, softmax over the 20, Z and atten.
// ---------------------------------------------------------------------------
__global__ __launch_bounds__(1024) void wmil_topk_out(
    const float* __restrict__ scores, const float* __restrict__ x,
    float* __restrict__ Z, float* __restrict__ atten)
{
    __shared__ float s_val[1024];
    __shared__ int   s_idx[1024];
    __shared__ int   s_thr[1024];
    __shared__ float sel_s[20];
    __shared__ int   sel_i[20];
    __shared__ float sel_w[20];

    const int tid = threadIdx.x;
    const int bag = blockIdx.x;
    const float* sc = scores + (size_t)bag * N_INST;

    float tv[10]; int ti[10];                        // top (descending)
    float lv[10]; int li[10];                        // bottom (ascending)
#pragma unroll
    for (int j = 0; j < 10; ++j) { tv[j] = -3.4e38f; ti[j] = 0; lv[j] = 3.4e38f; li[j] = 0; }

    for (int n = tid; n < N_INST; n += 1024) {
        float s = sc[n];
        if (s > tv[9]) {
            tv[9] = s; ti[9] = n;
#pragma unroll
            for (int j = 9; j > 0; --j)
                if (tv[j] > tv[j - 1]) {
                    float a = tv[j]; tv[j] = tv[j - 1]; tv[j - 1] = a;
                    int   b = ti[j]; ti[j] = ti[j - 1]; ti[j - 1] = b;
                }
        }
        if (s < lv[9]) {
            lv[9] = s; li[9] = n;
#pragma unroll
            for (int j = 9; j > 0; --j)
                if (lv[j] < lv[j - 1]) {
                    float a = lv[j]; lv[j] = lv[j - 1]; lv[j - 1] = a;
                    int   b = li[j]; li[j] = li[j - 1]; li[j - 1] = b;
                }
        }
    }

    int pt = 0, pb = 0;
    for (int round = 0; round < 10; ++round) {       // block argmax (top)
        s_val[tid] = (pt < 10) ? tv[pt] : -3.4e38f;
        s_idx[tid] = (pt < 10) ? ti[pt] : 0;
        s_thr[tid] = tid;
        __syncthreads();
        for (int st = 512; st > 0; st >>= 1) {
            if (tid < st && s_val[tid + st] > s_val[tid]) {
                s_val[tid] = s_val[tid + st];
                s_idx[tid] = s_idx[tid + st];
                s_thr[tid] = s_thr[tid + st];
            }
            __syncthreads();
        }
        if (tid == 0) { sel_s[round] = s_val[0]; sel_i[round] = s_idx[0]; }
        int winner = s_thr[0];
        __syncthreads();
        if (tid == winner) ++pt;
    }
    for (int round = 0; round < 10; ++round) {       // block argmin (bottom)
        s_val[tid] = (pb < 10) ? lv[pb] : 3.4e38f;
        s_idx[tid] = (pb < 10) ? li[pb] : 0;
        s_thr[tid] = tid;
        __syncthreads();
        for (int st = 512; st > 0; st >>= 1) {
            if (tid < st && s_val[tid + st] < s_val[tid]) {
                s_val[tid] = s_val[tid + st];
                s_idx[tid] = s_idx[tid + st];
                s_thr[tid] = s_thr[tid + st];
            }
            __syncthreads();
        }
        if (tid == 0) { sel_s[10 + round] = s_val[0]; sel_i[10 + round] = s_idx[0]; }
        int winner = s_thr[0];
        __syncthreads();
        if (tid == winner) ++pb;
    }

    if (tid == 0) {                                  // softmax over 20
        float m = sel_s[0];
        for (int j = 1; j < 20; ++j) m = fmaxf(m, sel_s[j]);
        float sum = 0.f;
        for (int j = 0; j < 20; ++j) { float e = __expf(sel_s[j] - m); sel_w[j] = e; sum += e; }
        float inv = 1.0f / sum;
        for (int j = 0; j < 20; ++j) sel_w[j] *= inv;
    }
    __syncthreads();

    float* at = atten + (size_t)bag * N_INST;        // race-free dense write
    for (int n = tid; n < N_INST; n += 1024) {
        float v = 0.f;
#pragma unroll
        for (int j = 0; j < 20; ++j) if (sel_i[j] == n) v = sel_w[j];
        at[n] = v;
    }

    {                                                // Z[bag, d], tid == d
        float acc = 0.f;
#pragma unroll
        for (int j = 0; j < 20; ++j)
            acc += sel_w[j] * x[((size_t)bag * N_INST + (size_t)sel_i[j]) * D_IN + tid];
        Z[bag * D_IN + tid] = acc;
    }
}

extern "C" void kernel_launch(void* const* d_in, const int* in_sizes, int n_in,
                              void* d_out, int out_size, void* d_ws, size_t ws_size,
                              hipStream_t stream) {
    const float* x  = (const float*)d_in[0];
    const float* Wv = (const float*)d_in[1];
    const float* bv = (const float*)d_in[2];
    const float* Wu = (const float*)d_in[3];
    const float* bu = (const float*)d_in[4];
    const float* Ww = (const float*)d_in[5];
    const float* bw = (const float*)d_in[6];

    float* Z     = (float*)d_out;                    // [8,1024]
    float* atten = (float*)d_out + B_BAGS * D_IN;    // [8,50000]

    float* scores = (float*)d_ws;                                        // 1.6 MB
    unsigned short* wpack =
        (unsigned short*)((char*)d_ws + (size_t)B_BAGS * N_INST * sizeof(float)); // 512 KB

    wmil_prepack<<<64, 256, 0, stream>>>(Wv, Wu, wpack);
    wmil_gemm_scores<<<dim3(GEMM_BLOCKS, B_BAGS), 256, 0, stream>>>(
        x, wpack, bv, bu, Ww, bw, scores);
    wmil_topk_out<<<B_BAGS, 1024, 0, stream>>>(scores, x, Z, atten);
}
